// MultiHeadAttention_4346506903689
// MI455X (gfx1250) — matrix operand order repaired and checked by reading, not compiled, not run on hardware
//
#include <hip/hip_runtime.h>
#include <hip/hip_bf16.h>
#include <stdint.h>

#define NH 12
#define BB 2
#define SS 2048
#define DD 768
#define DK 64

typedef __bf16 bf16_t;
typedef __attribute__((ext_vector_type(16))) __bf16 v16bf;
typedef __attribute__((ext_vector_type(8)))  __bf16 v8bf;
typedef __attribute__((ext_vector_type(8)))  float  v8f;
typedef __attribute__((ext_vector_type(4)))  float  v4f;

__device__ __forceinline__ v8f wmma_bf16(v16bf a, v16bf b, v8f c) {
  // D = A(16x32 bf16) * B(32x16 bf16) + C(16x16 f32)
  return __builtin_amdgcn_wmma_f32_16x16x32_bf16(
      false, a, false, b, (short)0, c, false, false);
}

// gfx1250 async copy: global -> LDS, 16 bytes per lane, tracked by ASYNCcnt.
__device__ __forceinline__ void async_copy_b128(const bf16_t* lds_dst,
                                                const bf16_t* gsrc) {
  const uint32_t lds = (uint32_t)(uintptr_t)lds_dst;  // low 32b = LDS byte addr
  asm volatile("global_load_async_to_lds_b128 %0, %1, off"
               :: "v"(lds), "v"((uint64_t)(uintptr_t)gsrc)
               : "memory");
}

// ---- Fragment loaders --------------------------------------------------
// A 16x32 (MxK), 16-bit: lanes 0-15 = M, half0 K{0..7,16..23}, half1 K{8..15,24..31}
__device__ __forceinline__ v16bf fragA_f32(const float* __restrict__ base,
                                           int ld, int row0, int k0) {
  const int l = threadIdx.x & 31;
  const float* p = base + (size_t)(row0 + (l & 15)) * ld + k0 + ((l >> 4) << 3);
  v4f x0 = *(const v4f*)(p);
  v4f x1 = *(const v4f*)(p + 4);
  v4f x2 = *(const v4f*)(p + 16);
  v4f x3 = *(const v4f*)(p + 20);
  v16bf f;
#pragma unroll
  for (int i = 0; i < 4; ++i) {
    f[i]      = (bf16_t)x0[i];
    f[4 + i]  = (bf16_t)x1[i];
    f[8 + i]  = (bf16_t)x2[i];
    f[12 + i] = (bf16_t)x3[i];
  }
  return f;
}

__device__ __forceinline__ v16bf fragA_bf16(const bf16_t* __restrict__ base,
                                            int ld, int row0, int k0) {
  const int l = threadIdx.x & 31;
  const bf16_t* p = base + (size_t)(row0 + (l & 15)) * ld + k0 + ((l >> 4) << 3);
  v8bf x0 = *(const v8bf*)p;
  v8bf x1 = *(const v8bf*)(p + 16);
  v16bf f;
#pragma unroll
  for (int i = 0; i < 8; ++i) { f[i] = x0[i]; f[8 + i] = x1[i]; }
  return f;
}

// B 32x16 (KxN), 16-bit, via B^T stored row-major (row = N, contiguous K):
// lanes 0-15: N=lane, K{0..7,16..23} ; lanes 16-31: N=lane-16, K{8..15,24..31}
__device__ __forceinline__ v16bf fragB_f32(const float* __restrict__ baseT,
                                           int ld, int col0, int k0) {
  const int l = threadIdx.x & 31;
  const float* p = baseT + (size_t)(col0 + (l & 15)) * ld + k0 + ((l >> 4) << 3);
  v16bf f;
#pragma unroll
  for (int c = 0; c < 4; ++c) {
    v4f x = *(const v4f*)(p + (c & 1) * 4 + (c >> 1) * 16);
#pragma unroll
    for (int i = 0; i < 4; ++i) f[4 * c + i] = (bf16_t)x[i];
  }
  return f;
}

__device__ __forceinline__ v16bf fragB_bf16(const bf16_t* baseT,
                                            int ld, int col0, int k0) {
  const int l = threadIdx.x & 31;
  const bf16_t* p = baseT + (size_t)(col0 + (l & 15)) * ld + k0 + ((l >> 4) << 3);
  v8bf x0 = *(const v8bf*)p;
  v8bf x1 = *(const v8bf*)(p + 16);
  v16bf f;
#pragma unroll
  for (int i = 0; i < 8; ++i) { f[i] = x0[i]; f[8 + i] = x1[i]; }
  return f;
}

// ---- QKV projection: out = X @ W^T, 32x32 tile per wave (2x2 WMMA) ----
__global__ void __launch_bounds__(128)
qkv_proj_kernel(const float* __restrict__ X, const float* __restrict__ W,
                bf16_t* __restrict__ out, int vtranspose) {
  const int wave = threadIdx.x >> 5;
  const int m0 = blockIdx.x * 32;
  const int n0 = (blockIdx.y * 4 + wave) * 32;
  v8f acc[2][2] = {};
  for (int k0 = 0; k0 < DD; k0 += 32) {
    v16bf a0 = fragA_f32(X, DD, m0,      k0);
    v16bf a1 = fragA_f32(X, DD, m0 + 16, k0);
    v16bf b0 = fragB_f32(W, DD, n0,      k0);
    v16bf b1 = fragB_f32(W, DD, n0 + 16, k0);
    acc[0][0] = wmma_bf16(a0, b0, acc[0][0]);
    acc[0][1] = wmma_bf16(a0, b1, acc[0][1]);
    acc[1][0] = wmma_bf16(a1, b0, acc[1][0]);
    acc[1][1] = wmma_bf16(a1, b1, acc[1][1]);
  }
  const int l = threadIdx.x & 31;
  const int rbase = (l >> 4) * 8;
#pragma unroll
  for (int mt = 0; mt < 2; ++mt)
#pragma unroll
    for (int nt = 0; nt < 2; ++nt) {
      const int n = n0 + nt * 16 + (l & 15);
      const int h = n >> 6, d = n & (DK - 1);
#pragma unroll
      for (int r = 0; r < 8; ++r) {
        const int m = m0 + mt * 16 + rbase + r;
        const int b_ = m >> 11, s = m & (SS - 1);
        const bf16_t val = (bf16_t)acc[mt][nt][r];
        if (!vtranspose)
          out[((size_t)(b_ * NH + h) * SS + s) * DK + d] = val;  // [B][H][S][dk]
        else
          out[((size_t)(b_ * NH + h) * DK + d) * SS + s] = val;  // [B][H][dk][S]
      }
    }
}

// ---- Flash attention: one wave = 32 queries, double-buffered async K/V ----
__global__ void __launch_bounds__(32)
attn_kernel(const bf16_t* __restrict__ q, const bf16_t* __restrict__ k,
            const bf16_t* __restrict__ vT, bf16_t* __restrict__ ctx) {
  __shared__ __align__(16) bf16_t kbuf[2][32][DK];  // 8 KB: 32 keys x 64
  __shared__ __align__(16) bf16_t vbuf[2][DK][32];  // 8 KB: 64 d x 32 keys
  __shared__ __align__(16) bf16_t pbuf[16][32];     // 1 KB: P transpose staging

  const int bh = blockIdx.y;
  const int q0 = blockIdx.x * 32;
  const bf16_t* qh = q  + (size_t)bh * SS * DK;
  const bf16_t* kh = k  + (size_t)bh * SS * DK;
  const bf16_t* vh = vT + (size_t)bh * DK * SS;

  const int l = threadIdx.x & 31;
  const int lane_n = l & 15;
  const int rbase  = (l >> 4) * 8;
  const float sc = 0.125f;  // 1/sqrt(64)

  v16bf aq[2][2];
#pragma unroll
  for (int t = 0; t < 2; ++t) {
    aq[t][0] = fragA_bf16(qh, DK, q0 + 16 * t, 0);
    aq[t][1] = fragA_bf16(qh, DK, q0 + 16 * t, 32);
  }

  v8f acc[2][4] = {};
  float mrun[2][8], lrun[2][8];
#pragma unroll
  for (int t = 0; t < 2; ++t)
#pragma unroll
    for (int r = 0; r < 8; ++r) { mrun[t][r] = -3.0e38f; lrun[t][r] = 0.f; }

  // Issue one 32-key block (K tile 4KB + V^T tile 4KB) = 16 async b128 ops.
  auto load_block = [&](int jb, int buf) {
    const int j0 = jb * 32;
#pragma unroll
    for (int i = 0; i < 8; ++i) {  // K: 32 rows x 128B
      const int idx = i * 32 + l;
      async_copy_b128((const bf16_t*)kbuf[buf] + idx * 8,
                      kh + (size_t)(j0 + (idx >> 3)) * DK + (idx & 7) * 8);
    }
#pragma unroll
    for (int i = 0; i < 8; ++i) {  // V^T: 64 rows x 64B
      const int idx = i * 32 + l;
      async_copy_b128((const bf16_t*)vbuf[buf] + idx * 8,
                      vh + (size_t)(idx >> 2) * SS + j0 + (idx & 3) * 8);
    }
  };

  const int nblocks = (q0 >> 5) + 1;  // causal: keys <= q0+31
  load_block(0, 0);

  for (int jb = 0; jb < nblocks; ++jb) {
    const int cur = jb & 1;
    const int j0 = jb * 32;
    if (jb + 1 < nblocks) {
      load_block(jb + 1, cur ^ 1);
      asm volatile("s_wait_asynccnt 0x10" ::: "memory");  // older block done
    } else {
      asm volatile("s_wait_asynccnt 0x0" ::: "memory");
    }
    const bf16_t* kb = (const bf16_t*)kbuf[cur];
    const bf16_t* vb = (const bf16_t*)vbuf[cur];

#pragma unroll
    for (int t = 0; t < 2; ++t) {
      const int qbase = q0 + 16 * t;
      if (j0 >= qbase + 16) continue;  // fully masked sub-block

      v8f s0 = {}, s1 = {};
      {
        v16bf b0 = fragB_bf16(kb, DK, 0, 0);
        v16bf b1 = fragB_bf16(kb, DK, 0, 32);
        s0 = wmma_bf16(aq[t][0], b0, s0);
        s0 = wmma_bf16(aq[t][1], b1, s0);
        v16bf b2 = fragB_bf16(kb, DK, 16, 0);
        v16bf b3 = fragB_bf16(kb, DK, 16, 32);
        s1 = wmma_bf16(aq[t][0], b2, s1);
        s1 = wmma_bf16(aq[t][1], b3, s1);
      }
      float p0[8], p1[8];
#pragma unroll
      for (int r = 0; r < 8; ++r) {
        const int m = qbase + rbase + r;
        float a_ = (j0 + lane_n      <= m) ? s0[r] * sc : -1e9f;
        float b_ = (j0 + 16 + lane_n <= m) ? s1[r] * sc : -1e9f;
        float tm = fmaxf(a_, b_);
#pragma unroll
        for (int xm = 1; xm < 16; xm <<= 1) tm = fmaxf(tm, __shfl_xor(tm, xm, 32));
        const float mnew  = fmaxf(mrun[t][r], tm);
        const float alpha = __expf(mrun[t][r] - mnew);
        mrun[t][r] = mnew;
        const float e0 = __expf(a_ - mnew);
        const float e1 = __expf(b_ - mnew);
        p0[r] = e0; p1[r] = e1;
        float rs = e0 + e1;
#pragma unroll
        for (int xm = 1; xm < 16; xm <<= 1) rs += __shfl_xor(rs, xm, 32);
        lrun[t][r] = lrun[t][r] * alpha + rs;
#pragma unroll
        for (int t4 = 0; t4 < 4; ++t4) acc[t][t4][r] *= alpha;
      }
      // C-layout -> A-layout transpose of P through LDS
      __syncthreads();
#pragma unroll
      for (int r = 0; r < 8; ++r) {
        pbuf[rbase + r][lane_n]      = (bf16_t)p0[r];
        pbuf[rbase + r][16 + lane_n] = (bf16_t)p1[r];
      }
      __syncthreads();
      v16bf pf;
      {
        const bf16_t* pp = &pbuf[0][0] + (l & 15) * 32 + ((l >> 4) << 3);
        v8bf x0 = *(const v8bf*)pp;
        v8bf x1 = *(const v8bf*)(pp + 16);
#pragma unroll
        for (int i = 0; i < 8; ++i) { pf[i] = x0[i]; pf[8 + i] = x1[i]; }
      }
#pragma unroll
      for (int t4 = 0; t4 < 4; ++t4) {
        v16bf bv = fragB_bf16(vb, 32, t4 * 16, 0);
        acc[t][t4] = wmma_bf16(pf, bv, acc[t][t4]);
      }
    }
  }

  const int b_ = bh / NH;
  const int h  = bh % NH;
#pragma unroll
  for (int t = 0; t < 2; ++t)
#pragma unroll
    for (int r = 0; r < 8; ++r) {
      const float inv = 1.0f / lrun[t][r];
      const int s = q0 + 16 * t + rbase + r;
      bf16_t* dst = ctx + ((size_t)b_ * SS + s) * DD + h * DK;
#pragma unroll
      for (int t4 = 0; t4 < 4; ++t4)
        dst[t4 * 16 + lane_n] = (bf16_t)(acc[t][t4][r] * inv);
    }
}

// ---- Output projection: out = ctx(bf16) @ Wo^T -> f32, 32x32 per wave ----
__global__ void __launch_bounds__(128)
out_proj_kernel(const bf16_t* __restrict__ Xc, const float* __restrict__ W,
                float* __restrict__ out) {
  const int wave = threadIdx.x >> 5;
  const int m0 = blockIdx.x * 32;
  const int n0 = (blockIdx.y * 4 + wave) * 32;
  v8f acc[2][2] = {};
  for (int k0 = 0; k0 < DD; k0 += 32) {
    v16bf a0 = fragA_bf16(Xc, DD, m0,      k0);
    v16bf a1 = fragA_bf16(Xc, DD, m0 + 16, k0);
    v16bf b0 = fragB_f32(W, DD, n0,      k0);
    v16bf b1 = fragB_f32(W, DD, n0 + 16, k0);
    acc[0][0] = wmma_bf16(a0, b0, acc[0][0]);
    acc[0][1] = wmma_bf16(a0, b1, acc[0][1]);
    acc[1][0] = wmma_bf16(a1, b0, acc[1][0]);
    acc[1][1] = wmma_bf16(a1, b1, acc[1][1]);
  }
  const int l = threadIdx.x & 31;
  const int rbase = (l >> 4) * 8;
#pragma unroll
  for (int mt = 0; mt < 2; ++mt)
#pragma unroll
    for (int nt = 0; nt < 2; ++nt) {
      const int n = n0 + nt * 16 + (l & 15);
#pragma unroll
      for (int r = 0; r < 8; ++r)
        out[(size_t)(m0 + mt * 16 + rbase + r) * DD + n] = acc[mt][nt][r];
    }
}

extern "C" void kernel_launch(void* const* d_in, const int* in_sizes, int n_in,
                              void* d_out, int out_size, void* d_ws, size_t ws_size,
                              hipStream_t stream) {
  const float* Q  = (const float*)d_in[0];
  const float* K  = (const float*)d_in[1];
  const float* V  = (const float*)d_in[2];
  // d_in[3] is the causal mask; applied analytically inside attn_kernel.
  const float* Wq = (const float*)d_in[4];
  const float* Wk = (const float*)d_in[5];
  const float* Wv = (const float*)d_in[6];
  const float* Wo = (const float*)d_in[7];

  const size_t HS = (size_t)BB * NH * SS * DK;  // 3,145,728 elems
  bf16_t* qws = (bf16_t*)d_ws;   // q  [B][H][S][dk]
  bf16_t* kws = qws + HS;        // k  [B][H][S][dk]
  bf16_t* vws = kws + HS;        // vT [B][H][dk][S]
  bf16_t* cws = vws + HS;        // ctx [B][S][D]
  (void)in_sizes; (void)n_in; (void)out_size; (void)ws_size;

  dim3 pg(BB * SS / 32, DD / 128);  // (128, 6), 128 threads = 4 waves (32x128 tile)
  qkv_proj_kernel<<<pg, 128, 0, stream>>>(Q, Wq, qws, 0);
  qkv_proj_kernel<<<pg, 128, 0, stream>>>(K, Wk, kws, 0);
  qkv_proj_kernel<<<pg, 128, 0, stream>>>(V, Wv, vws, 1);
  attn_kernel<<<dim3(SS / 32, BB * NH), 32, 0, stream>>>(qws, kws, vws, cws);
  out_proj_kernel<<<pg, 128, 0, stream>>>(cws, Wo, (float*)d_out);
}
